// RNNEncoderBasic_68298569940995
// MI455X (gfx1250) — compile-verified, matrix operand order checked by reading, not run on hardware
//
#include <hip/hip_runtime.h>
#include <hip/hip_bf16.h>
#include <math.h>

// Problem constants
#define BB 64
#define SS 256
#define EE 512
#define HH 512
#define G3 1536            // 3*H
#define NT 96              // G3/16 column tiles
#define KT 16              // 512/32 K tiles
#define PACKED_ELEMS 786432  // NT*KT*32*16 bf16 per matrix

typedef __attribute__((ext_vector_type(16))) __bf16 v16bf;
typedef __attribute__((ext_vector_type(8)))  __bf16 v8bf;
typedef __attribute__((ext_vector_type(8)))  float  v8f;

// ---- WMMA fragment loaders -------------------------------------------------
// A layout (bf16 16x32): lanes 0-15 hold rows M=0..15 with K = kk+{0..7},
// kk+{16..23}; lanes 16-31 hold K = kk+{8..15}, kk+{24..31}.  Caller passes
// base already offset by (lane>>4)*8 within the K chunk.
__device__ __forceinline__ v16bf load_a_bf16(const __bf16* p) {
  v8bf g0 = *(const v8bf*)(p);
  v8bf g1 = *(const v8bf*)(p + 16);
  v16bf a;
#pragma unroll
  for (int i = 0; i < 8; ++i) { a[i] = g0[i]; a[i + 8] = g1[i]; }
  return a;
}

__device__ __forceinline__ v16bf load_a_f32(const float* p) {
  v8f g0 = *(const v8f*)(p);
  v8f g1 = *(const v8f*)(p + 16);
  v16bf a;
#pragma unroll
  for (int i = 0; i < 8; ++i) { a[i] = (__bf16)g0[i]; a[i + 8] = (__bf16)g1[i]; }
  return a;
}

// ---- init: zero h double-buffers and grid-barrier counters ----------------
__global__ void init_kernel(float* __restrict__ hbuf, unsigned* __restrict__ bar) {
  int t = blockIdx.x * blockDim.x + threadIdx.x;
  if (t == 0) { bar[0] = 0u; bar[1] = 0u; }
  if (t < 4 * BB * HH) hbuf[t] = 0.0f;   // both parities, both directions
}

// ---- pack Wx_f/Wx_b/Wh_f/Wh_b (512x1536 f32 row-major) into WMMA-B layout -
// packed[(nt*16+kt)*32 + lane][16]: lane l -> column nt*16+(l&15), K-half l>>4
__global__ void pack_w_kernel(const float* __restrict__ Wx_f, const float* __restrict__ Wx_b,
                              const float* __restrict__ Wh_f, const float* __restrict__ Wh_b,
                              __bf16* __restrict__ wxp, __bf16* __restrict__ whp) {
  int t = blockIdx.x * blockDim.x + threadIdx.x;        // 4*96*16*32 = 196608
  if (t >= 4 * NT * KT * 32) return;
  int mat  = t / (NT * KT * 32);
  int rem  = t % (NT * KT * 32);
  int nt   = rem / (KT * 32);
  int rem2 = rem % (KT * 32);
  int kt   = rem2 >> 5;
  int ln   = rem2 & 31;
  const float* W = (mat == 0) ? Wx_f : (mat == 1) ? Wx_b : (mat == 2) ? Wh_f : Wh_b;
  __bf16* dst = ((mat < 2) ? (wxp + (size_t)mat * PACKED_ELEMS)
                           : (whp + (size_t)(mat - 2) * PACKED_ELEMS))
                + ((size_t)(nt * KT + kt) * 32 + ln) * 16;
  int n  = nt * 16 + (ln & 15);
  int kb = kt * 32 + (ln >> 4) * 8;
  v16bf o;
#pragma unroll
  for (int i = 0; i < 16; ++i) {
    int k = kb + (i & 7) + ((i >> 3) * 16);
    o[i] = (__bf16)W[(size_t)k * G3 + n];
  }
  *(v16bf*)dst = o;
}

// ---- gather embedding rows into xe[s*B+b][E] (bf16) ------------------------
__global__ void gather_embed_kernel(const int* __restrict__ src,
                                    const float* __restrict__ emb,
                                    __bf16* __restrict__ xe) {
  int t = blockIdx.x * blockDim.x + threadIdx.x;        // (S*B)*(E/8) = 1048576
  if (t >= SS * BB * (EE / 8)) return;
  int row = t >> 6;            // s*B + b
  int c   = t & 63;            // 8-elem chunk
  int s   = row >> 6;
  int b   = row & 63;
  int tok = src[b * SS + s];
  v8f v = *(const v8f*)(emb + (size_t)tok * EE + c * 8);
  v8bf o;
#pragma unroll
  for (int i = 0; i < 8; ++i) o[i] = (__bf16)v[i];
  *(v8bf*)(xe + (size_t)row * EE + c * 8) = o;
}

// ---- asp_g = asp_hidden @ Wa + bias  (tiny: 2 x 64 x 1536, K=512) ---------
__global__ void aspg_kernel(const float* __restrict__ asp,
                            const float* __restrict__ Wa_f, const float* __restrict__ b_f,
                            const float* __restrict__ Wa_b, const float* __restrict__ b_b,
                            float* __restrict__ aspg) {
  int t = blockIdx.x * blockDim.x + threadIdx.x;        // 196608
  if (t >= 2 * BB * G3) return;
  int dir = t / (BB * G3);
  int rem = t % (BB * G3);
  int b   = rem / G3;
  int g   = rem % G3;
  const float* Wa   = dir ? Wa_b : Wa_f;
  const float* bias = dir ? b_b : b_f;
  float acc = bias[g];
  const float* ap = asp + (size_t)b * 512;
  for (int k = 0; k < 512; ++k) acc += ap[k] * Wa[(size_t)k * G3 + g];
  aspg[t] = acc;
}

// ---- xg = xe @ Wx + asp_g  (both directions; backward uses reversed rows) -
// Block = (dir, ng of 48 columns, mg of 8 M-tiles).  The 48-column B panel
// (3 tiles x 16 kt = 48 KB) is staged in LDS once; all 8 waves share it and
// read fragments via ds_load_b128.  A (gathered embeddings, bf16) streams
// contiguously from global.  One wave -> 16 rows x 48 cols, 3 accumulators.
__global__ __launch_bounds__(256) void xg_gemm_kernel(const __bf16* __restrict__ xe,
                                                      const __bf16* __restrict__ wxp,
                                                      const float* __restrict__ aspg,
                                                      float* __restrict__ xg) {
  __shared__ __bf16 btile[3 * KT * 512];          // 48 KB B panel
  const int tid   = threadIdx.x;
  const int lane  = tid & 31;
  const int widx  = tid >> 5;                     // 0..7
  const int blk   = blockIdx.x;                   // 8192 blocks
  const int dir   = blk >> 12;                    // /4096
  const int rem   = blk & 4095;
  const int ng    = rem >> 7;                     // 0..31  (48-column group)
  const int mg    = rem & 127;                    // 0..127 (8 M-tile group)
  const int mtile = mg * 8 + widx;
  const int m0    = mtile * 16;
  const int rr    = m0 + (lane & 15);             // xg row = s*B + b
  const int s     = rr >> 6;
  const int b     = rr & 63;
  const int khalf = lane >> 4;
  const int srow  = (dir == 0) ? rr : ((SS - 1 - s) * BB + b);   // reversed seq

  // --- stage the block's B panel (columns ng*48 .. +47) into LDS ---
  {
    const __bf16* wsrc = wxp + (size_t)dir * PACKED_ELEMS + (size_t)(ng * 3) * (KT * 512);
    for (int c = tid; c < 3 * 512; c += 256) {    // 1536 chunks of 16 bf16 (24 KB panel *2)
      *(v16bf*)(btile + (size_t)c * 16) = *(const v16bf*)(wsrc + (size_t)c * 16);
    }
  }
  __syncthreads();

  const __bf16* abase = xe + (size_t)srow * EE + khalf * 8;
  const __bf16* bl    = btile + (size_t)lane * 16;
  v8f acc[3] = {v8f{}, v8f{}, v8f{}};
#pragma unroll 4
  for (int kt = 0; kt < KT; ++kt) {
    v16bf a = load_a_bf16(abase + kt * 32);
    v16bf b0 = *(const v16bf*)(bl + (size_t)(0 * KT + kt) * 512);
    v16bf b1 = *(const v16bf*)(bl + (size_t)(1 * KT + kt) * 512);
    v16bf b2 = *(const v16bf*)(bl + (size_t)(2 * KT + kt) * 512);
    acc[0] = __builtin_amdgcn_wmma_f32_16x16x32_bf16(false, a, false, b0, (short)0, acc[0], false, false);
    acc[1] = __builtin_amdgcn_wmma_f32_16x16x32_bf16(false, a, false, b1, (short)0, acc[1], false, false);
    acc[2] = __builtin_amdgcn_wmma_f32_16x16x32_bf16(false, a, false, b2, (short)0, acc[2], false, false);
  }
  float* xgd = xg + (size_t)dir * SS * BB * G3;
  const float* ag = aspg + (size_t)dir * BB * G3;
#pragma unroll
  for (int q = 0; q < 3; ++q) {
#pragma unroll
    for (int i = 0; i < 8; ++i) {               // D layout: lanes0-15 M=i, 16-31 M=8+i
      int m  = m0 + i + 8 * khalf;
      int bb = m & 63;
      int nc = (ng * 3 + q) * 16 + (lane & 15);
      xgd[(size_t)m * G3 + nc] = acc[q][i] + ag[(size_t)bb * G3 + nc];
    }
  }
}

// ---- software grid barrier (64 resident blocks, generation-counted) -------
__device__ __forceinline__ void grid_barrier(unsigned* bar, unsigned nblk, unsigned target) {
  __syncthreads();
  if (threadIdx.x == 0) {
    __threadfence();                               // release this step's stores
    unsigned prev = atomicAdd(&bar[0], 1u);
    if (prev == nblk - 1) {
      bar[0] = 0u;                                 // reset for next round
      __threadfence();
      atomicAdd(&bar[1], 1u);                      // publish generation
    } else {
      while (atomicAdd(&bar[1], 0u) < target) __builtin_amdgcn_s_sleep(8);
    }
    __threadfence();                               // acquire peers' stores
  }
  __syncthreads();
}

// ---- persistent recurrent scan: 64 blocks x 128 thr = 256 waves -----------
// block -> (dir = blockIdx>>5, jt = blockIdx&31); wave widx (0..3) -> mt.
// Wh gate tiles for this block's jt (48 KB) are staged in LDS ONCE before the
// timestep loop; inner loop reads B via ds_load_b128 instead of L2.
__global__ __launch_bounds__(128) void gru_scan_kernel(const float* __restrict__ xg,
                                                       const __bf16* __restrict__ whp,
                                                       float* __restrict__ hbuf,
                                                       float* __restrict__ out,
                                                       unsigned* __restrict__ bar) {
  __shared__ __bf16 wtile[3 * KT * 512];           // 3 gate tiles x 16 kt x 1KB = 48KB
  const int tid   = threadIdx.x;
  const int lane  = tid & 31;
  const int mt    = tid >> 5;                      // 0..3
  const int dir   = blockIdx.x >> 5;
  const int jt    = blockIdx.x & 31;
  const int khalf = lane >> 4;
  const int arow  = mt * 16 + (lane & 15);
  const int j     = jt * 16 + (lane & 15);

  // --- stage Wh gate tiles (jt, jt+32, jt+64) into LDS, one time ---
  {
    const __bf16* wsrc = whp + (size_t)dir * PACKED_ELEMS;
    for (int c = tid; c < 3 * 512; c += 128) {     // 1536 chunks of 16 bf16
      int g   = c / 512;                            // gate 0..2 (r,z,n)
      int r16 = c % 512;                            // 16-elem chunk within gate region
      *(v16bf*)(wtile + (size_t)g * (KT * 512) + (size_t)r16 * 16) =
          *(const v16bf*)(wsrc + (size_t)(jt + 32 * g) * (KT * 512) + (size_t)r16 * 16);
    }
  }
  __syncthreads();
  const __bf16* wl = wtile + (size_t)lane * 16;

  for (int s = 0; s < SS; ++s) {
    const float* hA = hbuf + (size_t)((s & 1) * 2 + dir) * (BB * HH);        // read
    float*       hB = hbuf + (size_t)(((s + 1) & 1) * 2 + dir) * (BB * HH);  // write
    const float* xr = xg + ((size_t)dir * SS + s) * (BB * G3);

    // Prefetch next step's xg rows (global_prefetch_b8) while GEMM runs.
    if (s + 1 < SS) {
      const float* xn = xr + (size_t)(BB * G3);
#pragma unroll
      for (int i = 0; i < 8; i += 2) {             // every other row covers the lines
        int m = mt * 16 + i + 8 * khalf;
        __builtin_prefetch(&xn[(size_t)m * G3 + j], 0, 3);
        __builtin_prefetch(&xn[(size_t)m * G3 + HH + j], 0, 3);
        __builtin_prefetch(&xn[(size_t)m * G3 + 2 * HH + j], 0, 3);
      }
    }

    v8f ar = {}, az = {}, an = {};
    const float* ab = hA + (size_t)arow * HH + khalf * 8;
#pragma unroll 4
    for (int kt = 0; kt < KT; ++kt) {
      v16bf a  = load_a_f32(ab + kt * 32);
      v16bf br = *(const v16bf*)(wl + (size_t)(0 * KT + kt) * 512);
      v16bf bz = *(const v16bf*)(wl + (size_t)(1 * KT + kt) * 512);
      v16bf bn = *(const v16bf*)(wl + (size_t)(2 * KT + kt) * 512);
      ar = __builtin_amdgcn_wmma_f32_16x16x32_bf16(false, a, false, br, (short)0, ar, false, false);
      az = __builtin_amdgcn_wmma_f32_16x16x32_bf16(false, a, false, bz, (short)0, az, false, false);
      an = __builtin_amdgcn_wmma_f32_16x16x32_bf16(false, a, false, bn, (short)0, an, false, false);
    }
#pragma unroll
    for (int i = 0; i < 8; ++i) {
      int m = mt * 16 + i + 8 * khalf;             // batch row
      float xgr = xr[(size_t)m * G3 + j];
      float xgz = xr[(size_t)m * G3 + HH + j];
      float xgn = xr[(size_t)m * G3 + 2 * HH + j];
      float hv  = hA[(size_t)m * HH + j];
      float r   = 1.0f / (1.0f + expf(-(xgr + ar[i])));
      float z   = 1.0f / (1.0f + expf(-(xgz + az[i])));
      float n   = tanhf(xgn + r * an[i]);
      float hn  = (1.0f - z) * hv + z * n;
      hB[(size_t)m * HH + j] = hn;
      if (dir == 0) out[((size_t)m * SS + s) * (2 * HH) + j] = hn;
      else          out[((size_t)m * SS + (SS - 1 - s)) * (2 * HH) + HH + j] = hn;
      if (s == SS - 1) {                            // encoder_last = [hT_b, hT_f]
        float* el = out + (size_t)BB * SS * 2 * HH;
        if (dir == 0) el[(size_t)m * (2 * HH) + HH + j] = hn;
        else          el[(size_t)m * (2 * HH) + j] = hn;
      }
    }
    grid_barrier(bar, 64u, (unsigned)(s + 1));
  }
}

// ---- launch ---------------------------------------------------------------
extern "C" void kernel_launch(void* const* d_in, const int* in_sizes, int n_in,
                              void* d_out, int out_size, void* d_ws, size_t ws_size,
                              hipStream_t stream) {
  const int*   src  = (const int*)d_in[0];
  // d_in[1] = src_lens (unused by reference math)
  const float* asp  = (const float*)d_in[2];
  const float* emb  = (const float*)d_in[3];
  const float* Wx_f = (const float*)d_in[4];
  const float* Wh_f = (const float*)d_in[5];
  const float* Wa_f = (const float*)d_in[6];
  const float* b_f  = (const float*)d_in[7];
  const float* Wx_b = (const float*)d_in[8];
  const float* Wh_b = (const float*)d_in[9];
  const float* Wa_b = (const float*)d_in[10];
  const float* b_b  = (const float*)d_in[11];
  float* out = (float*)d_out;

  // Workspace carve-up (~216 MB total)
  char* ws = (char*)d_ws;
  unsigned* bar  = (unsigned*)(ws);                         // 256 B
  __bf16*   xe   = (__bf16*)(ws + 256);                     // 16,777,216 B
  __bf16*   wxp  = (__bf16*)(ws + 256 + 16777216);          //  3,145,728 B
  __bf16*   whp  = (__bf16*)(ws + 256 + 16777216 + 3145728);//  3,145,728 B
  float*    aspg = (float*) (ws + 256 + 16777216 + 2 * 3145728);           //    786,432 B
  float*    xg   = (float*) (ws + 256 + 16777216 + 2 * 3145728 + 786432);  // 201,326,592 B
  float*    hbuf = (float*) (ws + 256 + 16777216 + 2 * 3145728 + 786432 + 201326592); // 524,288 B

  init_kernel<<<512, 256, 0, stream>>>(hbuf, bar);
  pack_w_kernel<<<768, 256, 0, stream>>>(Wx_f, Wx_b, Wh_f, Wh_b, wxp, whp);
  gather_embed_kernel<<<4096, 256, 0, stream>>>(src, emb, xe);
  aspg_kernel<<<768, 256, 0, stream>>>(asp, Wa_f, b_f, Wa_b, b_b, aspg);
  xg_gemm_kernel<<<8192, 256, 0, stream>>>(xe, wxp, aspg, xg);
  gru_scan_kernel<<<64, 128, 0, stream>>>(xg, whp, hbuf, out, bar);
}
